// SemiCurrSinkhornKnopp_stable_53626961658391
// MI455X (gfx1250) — matrix operand order repaired and checked by reading, not compiled, not run on hardware
//
#include <hip/hip_runtime.h>

typedef __bf16 bf16_t;
typedef __bf16 v16bf __attribute__((ext_vector_type(16)));
typedef __bf16 v8bf  __attribute__((ext_vector_type(8)));
typedef float  v8f   __attribute__((ext_vector_type(8)));

namespace sk {
constexpr int kNB = 4;          // batches
constexpr int kN = 2048;        // rows
constexpr int kK = 512;         // regular columns
constexpr int kKs = 513;        // columns + slack
constexpr int kWgPerB = 16;     // workgroups per batch
constexpr int kNWG = kNB * kWgPerB;   // 64 persistent workgroups
constexpr int kRows = kN / kWgPerB;   // 128 rows per workgroup
constexpr int kIters = 50;
constexpr float kInvEps   = 10.0f;                      // 1/EPSILON
constexpr float kFi       = 0.90909090909090909f;       // GAMMA/(GAMMA+EPS)
constexpr float kTiny     = 1.1920928955078125e-7f;     // finfo(f32).eps
constexpr float kStopErr  = 1e-6f;
constexpr float kStab     = 1e8f;
constexpr float kPa       = 1.0f / 2048.0f;
constexpr float kPbReg    = 0.5f / 512.0f;              // RHO/k
constexpr float kPbSlack  = 0.5f;                       // 1-RHO

// workspace layout (bytes)
constexpr size_t kEElems  = (size_t)kNB * kN * kK;      // 4,194,304
constexpr size_t kOffLse  = 0;                          // 8192 f32
constexpr size_t kOffEhi  = 32768;                      // bf16 hi part of E [b][i][j]
constexpr size_t kOffElo  = kOffEhi  + kEElems * 2;     // bf16 lo residual
constexpr size_t kOffEThi = kOffElo  + kEElems * 2;     // transposed E hi [b][j][i]
constexpr size_t kOffETlo = kOffEThi + kEElems * 2;     // transposed E lo
constexpr size_t kRedStride = 516;  // [0..511]=gy partial, [512]=amax, [513]=sum(z)
constexpr size_t kOffRed  = kOffETlo + kEElems * 2;
constexpr size_t kOffCnt  = kOffRed + (size_t)kIters * kNWG * kRedStride * 4;
}  // namespace sk

// ---------------- setup kernels ----------------

// one wave per row: lse = logsumexp over 512 logits
__global__ __launch_bounds__(256) void k_lse(const float* __restrict__ logits,
                                             void* wsv) {
  using namespace sk;
  float* lse = (float*)((char*)wsv + kOffLse);
  const int wv = threadIdx.x >> 5, lane = threadIdx.x & 31;
  const int row = blockIdx.x * 8 + wv;
  const float* p = logits + (size_t)row * kK;
  float v[16];
  float m = -3.4e38f;
#pragma unroll
  for (int t = 0; t < 16; ++t) { v[t] = p[lane + 32 * t]; m = fmaxf(m, v[t]); }
#pragma unroll
  for (int s = 16; s >= 1; s >>= 1) m = fmaxf(m, __shfl_xor(m, s, 32));
  float sum = 0.0f;
#pragma unroll
  for (int t = 0; t < 16; ++t) sum += expf(v[t] - m);
#pragma unroll
  for (int s = 16; s >= 1; s >>= 1) sum += __shfl_xor(sum, s, 32);
  if (lane == 0) lse[row] = m + logf(sum);
}

// E = exp((logit - lse)/eps) in split-bf16 (hi + exact residual lo),
// stored both row-major [b][i][j] and transposed [b][j][i] (one-time cost).
__global__ __launch_bounds__(256) void k_expE(const float* __restrict__ logits,
                                              void* wsv) {
  using namespace sk;
  const float* lse = (const float*)((char*)wsv + kOffLse);
  bf16_t* Ehi  = (bf16_t*)((char*)wsv + kOffEhi);
  bf16_t* Elo  = (bf16_t*)((char*)wsv + kOffElo);
  bf16_t* EThi = (bf16_t*)((char*)wsv + kOffEThi);
  bf16_t* ETlo = (bf16_t*)((char*)wsv + kOffETlo);
  const size_t idx = (size_t)blockIdx.x * 256 + threadIdx.x;
  const int row = (int)(idx >> 9);           // b*2048 + i
  const int b   = (int)(idx >> 20);
  const int i   = (int)((idx >> 9) & 2047);
  const int j   = (int)(idx & 511);
  float e = expf(kInvEps * (logits[idx] - lse[row]));   // in (0, 1]
  bf16_t h = (bf16_t)e;
  bf16_t l = (bf16_t)(e - (float)h);
  Ehi[idx] = h;
  Elo[idx] = l;
  const size_t tidx = ((size_t)(b * kK + j)) * kN + i;
  EThi[tidx] = h;
  ETlo[tidx] = l;
}

__global__ void k_zero_cnt(void* wsv) {
  int* cnt = (int*)((char*)wsv + sk::kOffCnt);
  if (threadIdx.x < 64) cnt[threadIdx.x] = 0;
}

// ---------------- persistent iteration kernel ----------------

__device__ __forceinline__ float blk_sum(float v, float* red) {
  red[threadIdx.x] = v;
  __syncthreads();
#pragma unroll
  for (int s = 128; s > 0; s >>= 1) {
    if ((int)threadIdx.x < s) red[threadIdx.x] += red[threadIdx.x + s];
    __syncthreads();
  }
  float r = red[0];
  __syncthreads();
  return r;
}

__device__ __forceinline__ float blk_max(float v, float* red) {
  red[threadIdx.x] = v;
  __syncthreads();
#pragma unroll
  for (int s = 128; s > 0; s >>= 1) {
    if ((int)threadIdx.x < s)
      red[threadIdx.x] = fmaxf(red[threadIdx.x], red[threadIdx.x + s]);
    __syncthreads();
  }
  float r = red[0];
  __syncthreads();
  return r;
}

__device__ __forceinline__ void grid_barrier(int* cnt, int it) {
  __syncthreads();
  if (threadIdx.x == 0) {
    __hip_atomic_fetch_add(&cnt[it], 1, __ATOMIC_RELEASE, __HIP_MEMORY_SCOPE_AGENT);
    while (__hip_atomic_load(&cnt[it], __ATOMIC_ACQUIRE, __HIP_MEMORY_SCOPE_AGENT) < sk::kNWG)
      __builtin_amdgcn_s_sleep(1);
  }
  __syncthreads();
  __threadfence();
}

__global__ __launch_bounds__(256) void k_sinkhorn(void* wsv, float* __restrict__ out) {
  using namespace sk;
  char* ws = (char*)wsv;
  const bf16_t* __restrict__ Ehi  = (const bf16_t*)(ws + kOffEhi);
  const bf16_t* __restrict__ Elo  = (const bf16_t*)(ws + kOffElo);
  const bf16_t* __restrict__ EThi = (const bf16_t*)(ws + kOffEThi);
  const bf16_t* __restrict__ ETlo = (const bf16_t*)(ws + kOffETlo);
  float* red_all = (float*)(ws + kOffRed);
  int* cnt = (int*)(ws + kOffCnt);

  const int tid = threadIdx.x;
  const int wg = blockIdx.x;
  const int myb = wg / kWgPerB;              // my batch
  const int i0 = (wg % kWgPerB) * kRows;     // my first row
  const int wv = tid >> 5;
  const int lane = tid & 31;
  const int mrow = lane & 15;
  const int hi = lane >> 4;

  // replicated per-batch state (identical in every WG -> deterministic scalars)
  __shared__ float st[kNB][kKs];      // v/eps
  __shared__ float sb[kNB][kKs];      // b
  __shared__ float slast[kNB][kKs];   // last_b
  __shared__ float sw[kNB][kK];       // w (semi weights)
  // per-WG row state
  __shared__ float s_s[kRows], s_es[kRows], s_a[kRows], s_gx[kRows], s_z[kRows];
  __shared__ __attribute__((aligned(64))) bf16_t s_yhi[kK];
  __shared__ __attribute__((aligned(64))) bf16_t s_ylo[kK];
  __shared__ __attribute__((aligned(64))) bf16_t s_zhi[kRows];
  __shared__ __attribute__((aligned(64))) bf16_t s_zlo[kRows];
  __shared__ float s_yslack;
  __shared__ float s_red[256];
  __shared__ float s_colf[kK];

  for (int idx = tid; idx < kNB * kKs; idx += 256) {
    const int bb = idx / kKs, j = idx % kKs;
    st[bb][j] = 0.0f;
    sb[bb][j] = 1.0f / (float)kKs;
    slast[bb][j] = 1.0f / (float)kKs;
  }
  for (int idx = tid; idx < kNB * kK; idx += 256) sw[idx / kK][idx % kK] = 1.0f;
  if (tid < kRows) { s_s[tid] = 0.0f; s_es[tid] = 1.0f; s_a[tid] = 0.0f; }
  __syncthreads();

  float err = 1.0f;
  bool stabled = false;

  for (int it = 0; it < kIters; ++it) {
    // y_j = e^{t_j} * b_j (my batch), split to bf16 hi/lo for WMMA
    for (int j = tid; j < kK; j += 256) {
      const float y = expf(st[myb][j]) * sb[myb][j];
      const bf16_t yh = (bf16_t)y;
      s_yhi[j] = yh;
      s_ylo[j] = (bf16_t)(y - (float)yh);
    }
    if (tid == 0) s_yslack = expf(st[myb][kK]) * sb[myb][kK];
    __syncthreads();

    // ---- Phase A: gx_i = sum_j E_ij * y_j  (split-bf16 WMMA row-sum GEMV) ----
    {
      const int r = i0 + wv * 16 + mrow;     // ISA A-layout: lane row = lane&15
      const bf16_t* ehr = Ehi + ((size_t)(myb * kN + r)) * kK;
      const bf16_t* elr = Elo + ((size_t)(myb * kN + r)) * kK;
      v8f c = {};
#pragma unroll 4
      for (int kt = 0; kt < kK / 32; ++kt) {
        const int c0 = kt * 32;
        __builtin_prefetch(ehr + c0 + 64, 0, 3);   // global_prefetch_b8 (near)
        __builtin_prefetch(elr + c0 + 64, 0, 3);
        // A 16x32 bf16 layout: elems0-15 at K=hi*16..hi*16+15 (matches B)
        v8bf ah0 = *(const v8bf*)(ehr + c0 + hi * 16);
        v8bf ah1 = *(const v8bf*)(ehr + c0 + hi * 16 + 8);
        v8bf al0 = *(const v8bf*)(elr + c0 + hi * 16);
        v8bf al1 = *(const v8bf*)(elr + c0 + hi * 16 + 8);
        v16bf Ah = __builtin_shufflevector(ah0, ah1, 0,1,2,3,4,5,6,7,8,9,10,11,12,13,14,15);
        v16bf Al = __builtin_shufflevector(al0, al1, 0,1,2,3,4,5,6,7,8,9,10,11,12,13,14,15);
        // B 32x16 broadcast: lanes0-15 K=c0+0..15, lanes16-31 K=c0+16..31
        v16bf Byh = *(const v16bf*)(&s_yhi[c0 + hi * 16]);
        v16bf Byl = *(const v16bf*)(&s_ylo[c0 + hi * 16]);
        c = __builtin_amdgcn_wmma_f32_16x16x32_bf16(false, Ah, false, Byh, (short)0, c, false, false);
        c = __builtin_amdgcn_wmma_f32_16x16x32_bf16(false, Ah, false, Byl, (short)0, c, false, false);
        c = __builtin_amdgcn_wmma_f32_16x16x32_bf16(false, Al, false, Byh, (short)0, c, false, false);
      }
      if (mrow == 0) {   // D layout: lane0 rows 0-7, lane16 rows 8-15 (all N equal)
#pragma unroll
        for (int rr = 0; rr < 8; ++rr) s_gx[wv * 16 + hi * 8 + rr] = c[rr];
      }
    }
    __syncthreads();

    // a_i = Pa / (e^{s_i}*(gx_i + yslack));  z_i = e^{s_i} * a_i (split to bf16)
    if (tid < kRows) {
      const float qb = s_es[tid] * (s_gx[tid] + s_yslack);
      const float a = kPa / qb;
      s_a[tid] = a;
      const float z = s_es[tid] * a;
      s_z[tid] = z;
      const bf16_t zh = (bf16_t)z;
      s_zhi[tid] = zh;
      s_zlo[tid] = (bf16_t)(z - (float)zh);
    }
    __syncthreads();
    const float amax = blk_max((tid < kRows) ? s_a[tid] : 0.0f, s_red);
    const float zsum = blk_sum((tid < kRows) ? s_z[tid] : 0.0f, s_red);  // slack colsum
    float* myred = red_all + ((size_t)it * kNWG + wg) * kRedStride;
    if (tid == 0) { myred[512] = amax; myred[513] = zsum; }
    __syncthreads();

    // ---- Phase B: gy_j partial = sum over my rows of E_ij * z_i
    //      (WMMA on transposed ET: A = z broadcast into row 0, B = ET tile) ----
#pragma unroll
    for (int g = 0; g < 4; ++g) {
      const int j0 = (wv * 4 + g) * 16;
      const bf16_t* thr = EThi + ((size_t)(myb * kK + j0 + mrow)) * kN + i0;
      const bf16_t* tlr = ETlo + ((size_t)(myb * kK + j0 + mrow)) * kN + i0;
      v8f c = {};
#pragma unroll
      for (int kt = 0; kt < kRows / 32; ++kt) {
        const int kb = kt * 32;
        __builtin_prefetch(thr + kb + 64, 0, 3);
        __builtin_prefetch(tlr + kb + 64, 0, 3);
        // A: only matrix row 0 nonzero (held by lanes 0 and 16 per ISA layout)
        v16bf Azh = {};
        v16bf Azl = {};
        if (mrow == 0) {
          v8bf z0 = *(const v8bf*)(&s_zhi[kb + hi * 16]);
          v8bf z1 = *(const v8bf*)(&s_zhi[kb + hi * 16 + 8]);
          Azh = __builtin_shufflevector(z0, z1, 0,1,2,3,4,5,6,7,8,9,10,11,12,13,14,15);
          v8bf l0 = *(const v8bf*)(&s_zlo[kb + hi * 16]);
          v8bf l1 = *(const v8bf*)(&s_zlo[kb + hi * 16 + 8]);
          Azl = __builtin_shufflevector(l0, l1, 0,1,2,3,4,5,6,7,8,9,10,11,12,13,14,15);
        }
        // B 32x16: lane n=lane&15 -> column j0+n; K chunk = kb + hi*16 .. +15
        v16bf Bh = *(const v16bf*)(thr + kb + hi * 16);
        v16bf Bl = *(const v16bf*)(tlr + kb + hi * 16);
        c = __builtin_amdgcn_wmma_f32_16x16x32_bf16(false, Azh, false, Bh, (short)0, c, false, false);
        c = __builtin_amdgcn_wmma_f32_16x16x32_bf16(false, Azl, false, Bh, (short)0, c, false, false);
        c = __builtin_amdgcn_wmma_f32_16x16x32_bf16(false, Azh, false, Bl, (short)0, c, false, false);
      }
      // D[0][n] lives in VGPR0 of lanes 0-15 (n = lane)
      if (hi == 0) myred[j0 + mrow] = c[0];
    }

    grid_barrier(cnt, it);   // cross-batch coupling point

    // ---- bn for ALL batches (replicated), err, stab ----
    float errp = 0.0f;
    float maxp = (tid < kNWG)
        ? red_all[((size_t)it * kNWG + tid) * kRedStride + 512] : 0.0f;  // fold a.max
    float bnv[kNB][2];
    float bnsl[kNB];
#pragma unroll
    for (int bb = 0; bb < kNB; ++bb) {
#pragma unroll
      for (int q = 0; q < 2; ++q) {
        const int j = tid + q * 256;
        float gy = 0.0f;
        for (int w = 0; w < kWgPerB; ++w)
          gy += red_all[((size_t)it * kNWG + bb * kWgPerB + w) * kRedStride + j];
        const float qta = expf(st[bb][j]) * gy;
        const float bn = powf(fmaxf(kPbReg / qta, kTiny), kFi) * sw[bb][j];
        bnv[bb][q] = bn;
        const float d = bn - slast[bb][j];
        errp += d * d;
        maxp = fmaxf(maxp, bn);
      }
      if (tid == 0) {  // slack column
        float gys = 0.0f;
        for (int w = 0; w < kWgPerB; ++w)
          gys += red_all[((size_t)it * kNWG + bb * kWgPerB + w) * kRedStride + 513];
        const float qta = expf(st[bb][kK]) * gys;
        const float bn = kPbSlack / qta;
        bnsl[bb] = bn;
        const float d = bn - slast[bb][kK];
        errp += d * d;
        maxp = fmaxf(maxp, bn);
      }
    }
    const float errsq = blk_sum(errp, s_red);
    const float gmax = blk_max(maxp, s_red);
    const bool stab = gmax > kStab;
    const float errn = sqrtf(errsq);

    // ---- updates (log-domain absorption when stab) ----
#pragma unroll
    for (int bb = 0; bb < kNB; ++bb) {
#pragma unroll
      for (int q = 0; q < 2; ++q) {
        const int j = tid + q * 256;
        const float bn = bnv[bb][q];
        if (stab) {
          st[bb][j] += logf(bn + kTiny);
          sw[bb][j] *= powf(fmaxf(bn, kTiny), kFi - 1.0f);
          sb[bb][j] = 1.0f;
        } else {
          sb[bb][j] = bn;
        }
        slast[bb][j] = bn;
      }
      if (tid == 0) {
        const float bn = bnsl[bb];
        if (stab) { st[bb][kK] += logf(bn + kTiny); sb[bb][kK] = 1.0f; }
        else      { sb[bb][kK] = bn; }
        slast[bb][kK] = bn;
      }
    }
    if (tid < kRows && stab) {
      s_s[tid] += logf(s_a[tid]);
      s_es[tid] = expf(s_s[tid]);
    }
    stabled = stabled || stab;
    err = errn;
    __syncthreads();
    if (err <= kStopErr) break;   // all WGs agree (bit-identical err)
  }

  // ---- plan = n * E * e^{s} * e^{t} * (stabled ? 1 : a*b), drop slack ----
  for (int j = tid; j < kK; j += 256)
    s_colf[j] = expf(st[myb][j]) * (stabled ? 1.0f : sb[myb][j]);
  __syncthreads();
  for (int ii = 0; ii < kRows; ++ii) {
    const int i = i0 + ii;
    const float rf = (float)kN * s_es[ii] * (stabled ? 1.0f : s_a[ii]);
    const bf16_t* eh = Ehi + ((size_t)(myb * kN + i)) * kK;
    const bf16_t* el = Elo + ((size_t)(myb * kN + i)) * kK;
    float* op = out + ((size_t)(myb * kN + i)) * kK;
#pragma unroll
    for (int q = 0; q < 2; ++q) {
      const int j = tid + q * 256;
      const float e = (float)eh[j] + (float)el[j];
      op[j] = e * rf * s_colf[j];
    }
  }
}

extern "C" void kernel_launch(void* const* d_in, const int* in_sizes, int n_in,
                              void* d_out, int out_size, void* d_ws, size_t ws_size,
                              hipStream_t stream) {
  (void)in_sizes; (void)n_in; (void)out_size; (void)ws_size;
  const float* logits = (const float*)d_in[0];
  float* out = (float*)d_out;

  k_zero_cnt<<<1, 64, 0, stream>>>(d_ws);                         // reset barriers
  k_lse<<<(sk::kNB * sk::kN) / 8, 256, 0, stream>>>(logits, d_ws);
  k_expE<<<(int)(sk::kEElems / 256), 256, 0, stream>>>(logits, d_ws);
  k_sinkhorn<<<sk::kNWG, 256, 0, stream>>>(d_ws, out);            // 64 persistent WGs
}
